// EnhancedGCN_3959959847629
// MI455X (gfx1250) — matrix operand order repaired and checked by reading, not compiled, not run on hardware
//
#include <hip/hip_runtime.h>
#include <hip/hip_bf16.h>

#define NN    100000
#define EE    1600000
#define GG    128
#define DD    128
#define HH    128
#define WALK  16
#define PED   16
#define LL    5
#define OUTD  10
#define BN_EPS 1e-5f

typedef __attribute__((ext_vector_type(16))) __bf16 v16bf;
typedef __attribute__((ext_vector_type(8)))  float  v8f;

// ---------------------------------------------------------------- utilities
__global__ void k_zero(float* __restrict__ p, int n) {
  int i = blockIdx.x * blockDim.x + threadIdx.x;
  for (; i < n; i += gridDim.x * blockDim.x) p[i] = 0.0f;
}

__global__ void k_copy(const float* __restrict__ s, float* __restrict__ d, int n) {
  int i = blockIdx.x * blockDim.x + threadIdx.x;
  if (i < n) d[i] = s[i];
}

__global__ void k_f2b(const float* __restrict__ s, __bf16* __restrict__ d, int n) {
  int i = blockIdx.x * blockDim.x + threadIdx.x;
  if (i < n) d[i] = (__bf16)s[i];
}

// weight convert + transpose: Wt[n*Kpad+k] = (n<Nout && k<K) ? W[k*Nout+n] : 0
__global__ void k_wt(const float* __restrict__ W, __bf16* __restrict__ Wt,
                     int K, int Nout, int Kpad, int Npad) {
  int i = blockIdx.x * blockDim.x + threadIdx.x;
  int total = Npad * Kpad;
  if (i >= total) return;
  int n = i / Kpad, k = i - n * Kpad;
  float v = (n < Nout && k < K) ? W[(size_t)k * Nout + n] : 0.0f;
  Wt[i] = (__bf16)v;
}

// ---------------------------------------------------------------- graph prep
__global__ void k_deg(const int* __restrict__ row0, const int* __restrict__ col0,
                      float* __restrict__ deg) {
  int e = blockIdx.x * blockDim.x + threadIdx.x;
  if (e >= EE + NN) return;
  int c = (e < EE) ? col0[e] : (e - EE);
  atomicAdd(&deg[c], 1.0f);
}

__global__ void k_dinv(const float* __restrict__ deg, float* __restrict__ dinv) {
  int i = blockIdx.x * blockDim.x + threadIdx.x;
  if (i >= NN) return;
  float d = deg[i];
  dinv[i] = (d > 0.0f) ? rsqrtf(fmaxf(d, 1.0f)) : 0.0f;
}

__global__ void k_norme(const int* __restrict__ row0, const int* __restrict__ col0,
                        const float* __restrict__ dinv, float* __restrict__ ne) {
  int e = blockIdx.x * blockDim.x + threadIdx.x;
  if (e >= EE + NN) return;
  int r = (e < EE) ? row0[e] : (e - EE);
  int c = (e < EE) ? col0[e] : (e - EE);
  ne[e] = dinv[r] * dinv[c];
}

__global__ void k_cnt(const int* __restrict__ batch, float* __restrict__ cnt) {
  int i = blockIdx.x * blockDim.x + threadIdx.x;
  if (i < NN) atomicAdd(&cnt[batch[i]], 1.0f);
}

__global__ void k_pinit(const int* __restrict__ batch, const float* __restrict__ cnt,
                        float* __restrict__ p) {
  int i = blockIdx.x * blockDim.x + threadIdx.x;
  if (i < NN) p[i] = 1.0f / fmaxf(cnt[batch[i]], 1.0f);
}

__global__ void k_rw_scatter(const int* __restrict__ row0, const int* __restrict__ col0,
                             const float* __restrict__ ne, const float* __restrict__ p,
                             float* __restrict__ pn) {
  int e = blockIdx.x * blockDim.x + threadIdx.x;
  if (e >= EE + NN) return;
  int r = (e < EE) ? row0[e] : (e - EE);
  int c = (e < EE) ? col0[e] : (e - EE);
  atomicAdd(&pn[c], p[r] * ne[e]);
}

__global__ void k_rw_update(float* __restrict__ p, const float* __restrict__ pn,
                            float* __restrict__ rwdst) {
  int i = blockIdx.x * blockDim.x + threadIdx.x;
  if (i >= NN) return;
  float pv = p[i];
  rwdst[i] = pv;                      // scan emits carry-in
  p[i] = 0.9f * pn[i] + 0.1f * pv;
}

// build padded projection input A[N x 160] bf16 = [emb | pe | 0-pad]
__global__ void k_build_projA(const int* __restrict__ x, const float* __restrict__ emb,
                              const float* __restrict__ rw, const float* __restrict__ pew,
                              const float* __restrict__ peb, __bf16* __restrict__ A) {
  int idx = blockIdx.x * blockDim.x + threadIdx.x;
  if (idx >= NN * 160) return;
  int i = idx / 160, d = idx - i * 160;
  float v;
  if (d < DD) {
    v = emb[(size_t)x[i] * DD + d];
  } else if (d < DD + PED) {
    int j = d - DD;
    float s = peb[j];
#pragma unroll
    for (int w = 0; w < WALK; ++w) s += rw[(size_t)w * NN + i] * pew[w * PED + j];
    v = s;
  } else {
    v = 0.0f;
  }
  A[idx] = (__bf16)v;
}

// ---------------------------------------------------------------- WMMA GEMM
// C[M x Nout] = act(A[M x K](bf16) @ Wt^T + bias), Wt is [Nout x K] bf16.
// Each wave owns a 64x16 output block: B fragments for the whole K are
// preloaded once into registers and reused across 4 row sub-tiles (4 WMMAs
// per B load).  256-thread blocks = 8 waves.  M must be a multiple of 16.
template <int ACT, int OUT_BF16, int K>
__global__ void k_gemm_wmma(const __bf16* __restrict__ A, const __bf16* __restrict__ Wt,
                            const float* __restrict__ bias, void* __restrict__ C,
                            int M, int Nout) {
  constexpr int KT = K / 32;
  const int lane = threadIdx.x & 31;
  const int wid  = threadIdx.x >> 5;
  const int tilesN  = Nout >> 4;
  const int tilesM  = (M + 63) >> 6;            // 64-row blocks
  int tile = blockIdx.x * (blockDim.x >> 5) + wid;
  if (tile >= tilesM * tilesN) return;          // wave-uniform exit
  const int tm = tile / tilesN;
  const int tn = tile - tm * tilesN;
  const int sel = lane >> 4;                    // 0 = lanes 0-15, 1 = lanes 16-31
  const int hr  = lane & 15;

  // ---- preload all B fragments (lane n holds col n, K = sel*8 + j (+16 for j>=8))
  const __bf16* bp = Wt + (size_t)(tn * 16 + hr) * K + sel * 8;
  v16bf bfrag[KT];
#pragma unroll
  for (int kt = 0; kt < KT; ++kt) {
#pragma unroll
    for (int j = 0; j < 16; ++j) bfrag[kt][j] = bp[kt * 32 + j + (j & 8)];
  }

  const int col = tn * 16 + hr;
  const float bv = bias[col];

#pragma unroll
  for (int sub = 0; sub < 4; ++sub) {
    const int m0 = tm * 64 + sub * 16;
    if (m0 >= M) break;                         // wave-uniform (M % 16 == 0)
    const __bf16* ap = A + (size_t)(m0 + hr) * K + sel * 8;
    v8f acc = {};
#pragma unroll
    for (int kt = 0; kt < KT; ++kt) {
      v16bf a;
#pragma unroll
      for (int i = 0; i < 8; ++i) {             // A: K = kt*32 + sel*8 + i (+16)
        a[i]     = ap[kt * 32 + i];
        a[i + 8] = ap[kt * 32 + 16 + i];
      }
      acc = __builtin_amdgcn_wmma_f32_16x16x32_bf16(false, a, false, bfrag[kt],
                                                    (short)0, acc, false, false);
    }
#pragma unroll
    for (int i = 0; i < 8; ++i) {
      int m = m0 + sel * 8 + i;                 // C: M = i + 8*sel
      float v = acc[i] + bv;
      if (ACT) v = fmaxf(v, 0.0f);
      if (OUT_BF16) ((__bf16*)C)[(size_t)m * Nout + col] = (__bf16)v;
      else          ((float*)C)[(size_t)m * Nout + col]  = v;
    }
  }
}

// ---------------------------------------------------------------- sparse agg
// one wave per edge; lane moves float4 -> 32*16B = full 128-float row
__global__ void k_edge_agg(const int* __restrict__ row0, const int* __restrict__ col0,
                           const float* __restrict__ h, float* __restrict__ agg) {
  int idx = blockIdx.x * blockDim.x + threadIdx.x;
  int e = idx >> 5;
  if (e >= EE) return;
  int lane = idx & 31;
  int r = row0[e], c = col0[e];
  const float4 v = *((const float4*)(h + (size_t)r * HH) + lane);
  float* dst = agg + (size_t)c * HH + lane * 4;
  atomicAdd(dst + 0, v.x);
  atomicAdd(dst + 1, v.y);
  atomicAdd(dst + 2, v.z);
  atomicAdd(dst + 3, v.w);
}

// ---------------------------------------------------------------- batchnorm
__global__ void k_bn_stats(const float* __restrict__ z, float* __restrict__ sum,
                           float* __restrict__ sumsq) {
  int c = threadIdx.x;                          // 128 channels
  int r0 = blockIdx.x * 512;
  int r1 = r0 + 512; if (r1 > NN) r1 = NN;
  float s = 0.0f, s2 = 0.0f;
  for (int r = r0; r < r1; ++r) {
    float v = z[(size_t)r * HH + c];
    s += v; s2 += v * v;
  }
  atomicAdd(&sum[c], s);
  atomicAdd(&sumsq[c], s2);
}

__global__ void k_bn_fin(const float* __restrict__ sum, const float* __restrict__ sumsq,
                         float* __restrict__ mu, float* __restrict__ inv) {
  int c = threadIdx.x;
  float m = sum[c] / (float)NN;
  float var = fmaxf(sumsq[c] / (float)NN - m * m, 0.0f);
  mu[c] = m;
  inv[c] = rsqrtf(var + BN_EPS);
}

// h = relu(BN(z)) + h ; hb = bf16(h)
__global__ void k_gin_post(const float* __restrict__ z, const float* __restrict__ mu,
                           const float* __restrict__ inv, const float* __restrict__ g,
                           const float* __restrict__ b, float* __restrict__ h,
                           __bf16* __restrict__ hb) {
  int i = blockIdx.x * blockDim.x + threadIdx.x;
  if (i >= NN * HH) return;
  int c = i & (HH - 1);
  float v = (z[i] - mu[c]) * inv[c] * g[c] + b[c];
  v = fmaxf(v, 0.0f) + h[i];
  h[i] = v;
  hb[i] = (__bf16)v;
}

__global__ void k_add(float* __restrict__ t, const float* __restrict__ h, int n) {
  int i = blockIdx.x * blockDim.x + threadIdx.x;
  if (i < n) t[i] += h[i];
}

// h = BN(t) (no relu)
__global__ void k_ffn_post(const float* __restrict__ t, const float* __restrict__ mu,
                           const float* __restrict__ inv, const float* __restrict__ g,
                           const float* __restrict__ b, float* __restrict__ h) {
  int i = blockIdx.x * blockDim.x + threadIdx.x;
  if (i >= NN * HH) return;
  int c = i & (HH - 1);
  h[i] = (t[i] - mu[c]) * inv[c] * g[c] + b[c];
}

// ---------------------------------------------------------------- pooling/head
__global__ void k_pool(const int* __restrict__ batch, const float* __restrict__ h,
                       float* __restrict__ gp) {
  int idx = blockIdx.x * blockDim.x + threadIdx.x;
  int nid = idx >> 5;
  if (nid >= NN) return;
  int lane = idx & 31;
  int b = batch[nid];
  const float4 v = *((const float4*)(h + (size_t)nid * HH) + lane);
  float* dst = gp + (size_t)b * HH + lane * 4;
  atomicAdd(dst + 0, v.x);
  atomicAdd(dst + 1, v.y);
  atomicAdd(dst + 2, v.z);
  atomicAdd(dst + 3, v.w);
}

__global__ void k_pool_div(const float* __restrict__ gp, const float* __restrict__ cnt,
                           __bf16* __restrict__ gbf) {
  int i = blockIdx.x * blockDim.x + threadIdx.x;
  if (i >= GG * HH) return;
  int g = i >> 7;
  gbf[i] = (__bf16)(gp[i] / fmaxf(cnt[g], 1.0f));
}

__global__ void k_head_out(const __bf16* __restrict__ ghb, const float* __restrict__ w2,
                           const float* __restrict__ b2, float* __restrict__ out) {
  int i = blockIdx.x * blockDim.x + threadIdx.x;
  if (i >= GG * OUTD) return;
  int g = i / OUTD, o = i - g * OUTD;
  float s = b2[o];
  for (int k = 0; k < HH; ++k)
    s += (float)ghb[g * HH + k] * w2[k * OUTD + o];
  out[i] = s;
}

// ---------------------------------------------------------------- launcher
extern "C" void kernel_launch(void* const* d_in, const int* in_sizes, int n_in,
                              void* d_out, int out_size, void* d_ws, size_t ws_size,
                              hipStream_t stream) {
  const int*   x      = (const int*)d_in[0];
  const int*   ei     = (const int*)d_in[1];
  const int*   batch  = (const int*)d_in[2];
  const float* emb    = (const float*)d_in[3];
  const float* pe_w   = (const float*)d_in[4];
  const float* pe_b   = (const float*)d_in[5];
  const float* proj_w = (const float*)d_in[6];
  const float* proj_b = (const float*)d_in[7];
  const float* gin_w1 = (const float*)d_in[8];
  const float* gin_b1 = (const float*)d_in[9];
  const float* gin_w2 = (const float*)d_in[10];
  const float* gin_b2 = (const float*)d_in[11];
  const float* bn_g   = (const float*)d_in[12];
  const float* bn_b   = (const float*)d_in[13];
  const float* ffn_w1 = (const float*)d_in[14];
  const float* ffn_b1 = (const float*)d_in[15];
  const float* ffn_w2 = (const float*)d_in[16];
  const float* ffn_b2 = (const float*)d_in[17];
  const float* fbn_g  = (const float*)d_in[18];
  const float* fbn_b  = (const float*)d_in[19];
  const float* out_w1 = (const float*)d_in[20];
  const float* out_b1 = (const float*)d_in[21];
  const float* out_w2 = (const float*)d_in[22];
  const float* out_b2 = (const float*)d_in[23];

  const int* row0 = ei;
  const int* col0 = ei + EE;

  // ---- workspace carve-up
  char* wsb = (char*)d_ws;
  size_t off = 0;
  auto take = [&](size_t bytes) -> char* {
    char* p = wsb + off;
    off += (bytes + 255) & ~(size_t)255;
    return p;
  };
  float*  deg   = (float*)take((size_t)NN * 4);
  float*  dinv  = (float*)take((size_t)NN * 4);
  float*  ne    = (float*)take((size_t)(EE + NN) * 4);
  float*  cnt   = (float*)take((size_t)GG * 4);
  float*  p     = (float*)take((size_t)NN * 4);
  float*  pn    = (float*)take((size_t)NN * 4);
  float*  rw    = (float*)take((size_t)WALK * NN * 4);
  float*  h     = (float*)take((size_t)NN * HH * 4);
  float*  aggf  = (float*)take((size_t)NN * HH * 4);      // agg / z / t scratch
  __bf16* buf1  = (__bf16*)take((size_t)NN * HH * 2);     // bf16 activations
  __bf16* buf2  = (__bf16*)take((size_t)NN * 2 * HH * 2); // bf16 wide scratch
  float*  stats = (float*)take(4 * HH * 4);               // sum|sumsq|mu|inv
  float*  gp    = (float*)take((size_t)GG * HH * 4);
  __bf16* gbf   = (__bf16*)take((size_t)GG * HH * 2);
  __bf16* ghb   = (__bf16*)take((size_t)GG * HH * 2);
  __bf16* wt_proj = (__bf16*)take(128 * 160 * 2);
  __bf16* w1t[LL], *w2t[LL], *f1t[LL], *f2t[LL];
  for (int l = 0; l < LL; ++l) {
    w1t[l] = (__bf16*)take(128 * 128 * 2);
    w2t[l] = (__bf16*)take(128 * 128 * 2);
    f1t[l] = (__bf16*)take(256 * 128 * 2);
    f2t[l] = (__bf16*)take(128 * 256 * 2);
  }
  __bf16* out1t = (__bf16*)take(128 * 128 * 2);

  float* bsum = stats, *bsq = stats + HH, *bmu = stats + 2 * HH, *binv = stats + 3 * HH;

  const int T = 256;
  auto gr = [](long n, int t) { return (int)((n + t - 1) / t); };
  // wave-tiles (64 rows x 16 cols each) -> blocks of 8 waves
  auto gemm_blocks = [](int M, int Nout) {
    long tiles = (long)((M + 63) / 64) * (Nout / 16);
    return (int)((tiles + 7) / 8);
  };

  // ---- weight convert + transpose (bf16, [Nout x Kpad])
  k_wt<<<gr(128 * 160, T), T, 0, stream>>>(proj_w, wt_proj, 144, 128, 160, 128);
  for (int l = 0; l < LL; ++l) {
    k_wt<<<gr(128 * 128, T), T, 0, stream>>>(gin_w1 + (size_t)l * HH * HH, w1t[l], 128, 128, 128, 128);
    k_wt<<<gr(128 * 128, T), T, 0, stream>>>(gin_w2 + (size_t)l * HH * HH, w2t[l], 128, 128, 128, 128);
    k_wt<<<gr(256 * 128, T), T, 0, stream>>>(ffn_w1 + (size_t)l * HH * 2 * HH, f1t[l], 128, 256, 128, 256);
    k_wt<<<gr(128 * 256, T), T, 0, stream>>>(ffn_w2 + (size_t)l * 2 * HH * HH, f2t[l], 256, 128, 256, 128);
  }
  k_wt<<<gr(128 * 128, T), T, 0, stream>>>(out_w1, out1t, 128, 128, 128, 128);

  // ---- graph normalization + per-graph counts
  k_zero<<<gr(NN, T), T, 0, stream>>>(deg, NN);
  k_deg<<<gr(EE + NN, T), T, 0, stream>>>(row0, col0, deg);
  k_dinv<<<gr(NN, T), T, 0, stream>>>(deg, dinv);
  k_norme<<<gr(EE + NN, T), T, 0, stream>>>(row0, col0, dinv, ne);
  k_zero<<<gr(GG, T), T, 0, stream>>>(cnt, GG);
  k_cnt<<<gr(NN, T), T, 0, stream>>>(batch, cnt);
  k_pinit<<<gr(NN, T), T, 0, stream>>>(batch, cnt, p);

  // ---- random-walk positional encoding (16 scan steps)
  for (int s = 0; s < WALK; ++s) {
    k_zero<<<gr(NN, T), T, 0, stream>>>(pn, NN);
    k_rw_scatter<<<gr(EE + NN, T), T, 0, stream>>>(row0, col0, ne, p, pn);
    k_rw_update<<<gr(NN, T), T, 0, stream>>>(p, pn, rw + (size_t)s * NN);
  }

  // ---- input projection: [N x 160] bf16 @ Wt -> h [N x 128] f32
  k_build_projA<<<gr((long)NN * 160, T), T, 0, stream>>>(x, emb, rw, pe_w, pe_b, buf2);
  k_gemm_wmma<0, 0, 160><<<gemm_blocks(NN, 128), T, 0, stream>>>(buf2, wt_proj, proj_b, h, NN, 128);

  auto run_bn = [&](const float* z) {
    k_zero<<<1, T, 0, stream>>>(stats, 2 * HH);
    k_bn_stats<<<gr(NN, 512), HH, 0, stream>>>(z, bsum, bsq);
    k_bn_fin<<<1, HH, 0, stream>>>(bsum, bsq, bmu, binv);
  };

  // ---- stacked GIN + FFN layers
  for (int l = 0; l < LL; ++l) {
    // GIN aggregate: agg = h + scatter_add(h[row] -> col)
    k_copy<<<gr((long)NN * HH, T), T, 0, stream>>>(h, aggf, NN * HH);
    k_edge_agg<<<gr((long)EE * 32, T), T, 0, stream>>>(row0, col0, h, aggf);
    k_f2b<<<gr((long)NN * HH, T), T, 0, stream>>>(aggf, buf1, NN * HH);
    // z = relu(agg@W1+b1)@W2+b2
    k_gemm_wmma<1, 1, 128><<<gemm_blocks(NN, 128), T, 0, stream>>>(buf1, w1t[l], gin_b1 + l * HH, buf2, NN, 128);
    k_gemm_wmma<0, 0, 128><<<gemm_blocks(NN, 128), T, 0, stream>>>(buf2, w2t[l], gin_b2 + l * HH, aggf, NN, 128);
    run_bn(aggf);
    k_gin_post<<<gr((long)NN * HH, T), T, 0, stream>>>(aggf, bmu, binv, bn_g + l * HH, bn_b + l * HH, h, buf1);
    // FFN: f = relu(h@Wf1+b)@Wf2+b ; h = BN(f + h)
    k_gemm_wmma<1, 1, 128><<<gemm_blocks(NN, 256), T, 0, stream>>>(buf1, f1t[l], ffn_b1 + l * 2 * HH, buf2, NN, 256);
    k_gemm_wmma<0, 0, 256><<<gemm_blocks(NN, 128), T, 0, stream>>>(buf2, f2t[l], ffn_b2 + l * HH, aggf, NN, 128);
    k_add<<<gr((long)NN * HH, T), T, 0, stream>>>(aggf, h, NN * HH);
    run_bn(aggf);
    k_ffn_post<<<gr((long)NN * HH, T), T, 0, stream>>>(aggf, bmu, binv, fbn_g + l * HH, fbn_b + l * HH, h);
  }

  // ---- global mean pool + head
  k_zero<<<gr(GG * HH, T), T, 0, stream>>>(gp, GG * HH);
  k_pool<<<gr((long)NN * 32, T), T, 0, stream>>>(batch, h, gp);
  k_pool_div<<<gr(GG * HH, T), T, 0, stream>>>(gp, cnt, gbf);
  k_gemm_wmma<1, 1, 128><<<gemm_blocks(GG, 128), T, 0, stream>>>(gbf, out1t, out_b1, ghb, GG, 128);
  k_head_out<<<gr(GG * OUTD, T), T, 0, stream>>>(ghb, out_w2, out_b2, (float*)d_out);
}
